// SEGGRU_962072674356
// MI455X (gfx1250) — matrix operand order repaired and checked by reading, not compiled, not run on hardware
//
#include <hip/hip_runtime.h>
#include <cstdint>
#include <cstddef>

// ---------------------------------------------------------------------------
// Types for CDNA5 WMMA (wave32): v_wmma_f32_16x16x32_bf16
// ---------------------------------------------------------------------------
typedef __attribute__((ext_vector_type(16))) __bf16 bf16x16;
typedef __attribute__((ext_vector_type(8)))  __bf16 bf16x8;
typedef __attribute__((ext_vector_type(2)))  __bf16 bf16x2;
typedef __attribute__((ext_vector_type(8)))  float  f32x8;
typedef __attribute__((ext_vector_type(2)))  float  f32x2;

// Problem constants (match reference)
#define BB   512
#define WS   128
#define II   256
#define HH   512
#define NWS  64
#define PRE  24
#define G3H  1536   // 3*H

// ---------------------------------------------------------------------------
// Helpers  (native hardware conversions; backend picks v_cvt_*bf16*)
// ---------------------------------------------------------------------------
__device__ __forceinline__ __bf16 f2bf(float f) { return (__bf16)f; }
__device__ __forceinline__ float  bf2f(__bf16 b) { return (float)b; }

__device__ __forceinline__ float sigmoid_f(float x) {
  return 1.0f / (1.0f + __expf(-x));
}
__device__ __forceinline__ float tanh_f(float x) {
  float e = __expf(-2.0f * x);
  return (1.0f - e) / (1.0f + e);
}

// A-fragment: 16x32 (MxK) bf16 tile, row-major source with leading dim ld.
// Layout: lanes 0-15 row M=lane hold K {0..15}; lanes 16-31 same
// rows hold K {16..31}. Two contiguous 16B chunks per lane.
__device__ __forceinline__ bf16x16 load_a_frag(const __bf16* A, int ld, int lane) {
  int m = lane & 15, half = lane >> 4;
  const __bf16* p = A + m * ld + 16 * half;
  bf16x8 lo = *(const bf16x8*)(p);
  bf16x8 hi = *(const bf16x8*)(p + 8);
  return __builtin_shufflevector(lo, hi, 0,1,2,3,4,5,6,7,8,9,10,11,12,13,14,15);
}

// B-fragment: 32x16 (KxN) bf16 tile, sourced from an N-major ("B^T") matrix
// Bt[n][k] with leading dim ld. Lane n holds K {16*half .. 16*half+15}:
// one contiguous 32B load per lane.
__device__ __forceinline__ bf16x16 load_b_frag(const __bf16* Bt, int ld, int lane) {
  int n = lane & 15, half = lane >> 4;
  return *(const bf16x16*)(Bt + n * ld + 16 * half);
}

__device__ __forceinline__ f32x8 wmma_bf16(bf16x16 a, bf16x16 b, f32x8 c) {
  return __builtin_amdgcn_wmma_f32_16x16x32_bf16(false, a, false, b, (short)0, c,
                                                 false, false);
}

// ---------------------------------------------------------------------------
// Kernel: f32 -> bf16 convert (weights)
// ---------------------------------------------------------------------------
__global__ __launch_bounds__(256) void cvt_f32_bf16(const float* __restrict__ s,
                                                    __bf16* __restrict__ d, int n) {
  int i = (blockIdx.x * 256 + threadIdx.x) * 2;   // pairs -> packed cvt + b32 store
  if (i < n) {
    bf16x2 p;
    p[0] = f2bf(s[i]);
    p[1] = f2bf(s[i + 1]);
    *(bf16x2*)(d + i) = p;
  }
}

// ---------------------------------------------------------------------------
// Kernel: transpose+convert x [B][WS][I] f32 -> xT [B][I][WS] bf16
// (makes x a contiguous N-major B-matrix for the fc1 GEMM)
// ---------------------------------------------------------------------------
__global__ __launch_bounds__(256) void transpose_x(const float* __restrict__ x,
                                                   __bf16* __restrict__ xT) {
  __shared__ float tile[32][33];
  int b  = blockIdx.x;        // batch
  int tt = blockIdx.y;        // t-tile  (WS/32 = 4)
  int it = blockIdx.z;        // i-tile  (I/32  = 8)
  int tx = threadIdx.x & 31, ty = threadIdx.x >> 5;   // 32 x 8
  const float* xb = x + (size_t)b * WS * II;
#pragma unroll
  for (int r = 0; r < 32; r += 8)
    tile[ty + r][tx] = xb[(tt * 32 + ty + r) * II + it * 32 + tx];
  __syncthreads();
  __bf16* xTb = xT + (size_t)b * II * WS;
#pragma unroll
  for (int r = 0; r < 32; r += 8)
    xTb[(it * 32 + ty + r) * WS + tt * 32 + tx] = f2bf(tile[tx][ty + r]);
}

// ---------------------------------------------------------------------------
// Kernel: fused fc1 + input-side gate projection.
//   Per batch b (one WG, 8 waves):
//     Y_b[64,256]  = fc1_w(64x128) @ x_b(128x256) + fc1_b   -> LDS (bf16)
//     gx_b[64,1536]= Y_b @ W_ih^T + b_ih                    -> global (bf16)
// ---------------------------------------------------------------------------
__global__ __launch_bounds__(256) void fused_fc1_gx(
    const __bf16* __restrict__ xT,    // [B][I][WS]  (N-major B for fc1)
    const __bf16* __restrict__ fc1w,  // [64][128]   (row-major A)
    const float*  __restrict__ fc1b,  // [64]
    const __bf16* __restrict__ Wih,   // [1536][256] (N-major B for gx)
    const float*  __restrict__ bih,   // [1536]
    __bf16* __restrict__ gx)          // [B][64][1536]
{
  __shared__ __align__(32) __bf16 Y[NWS][II];   // 32 KB
  int b = blockIdx.x;
  int tid = threadIdx.x, wave = tid >> 5, lane = tid & 31;
  int n = lane & 15, half = lane >> 4;
  const __bf16* xTb = xT + (size_t)b * II * WS;

  // Phase 1: Y = fc1_w @ x_b   (M=64, N=256, K=128).
  // Wave w owns tiles id = w*8..w*8+7  -> mt = w/2 constant per wave.
  {
    int mt = wave >> 1;
    bf16x16 afr[4];                      // hoisted A fragments (K=128)
#pragma unroll
    for (int k = 0; k < 4; ++k)
      afr[k] = load_a_frag(fc1w + mt * 16 * WS + k * 32, WS, lane);
    for (int i = 0; i < 8; ++i) {
      int nt = (wave * 8 + i) & 15;
      f32x8 c = {};
#pragma unroll
      for (int k = 0; k < 4; ++k)
        c = wmma_bf16(afr[k], load_b_frag(xTb + nt * 16 * WS + k * 32, WS, lane), c);
#pragma unroll
      for (int v = 0; v < 8; ++v) {
        int row = mt * 16 + half * 8 + v;
        Y[row][nt * 16 + n] = f2bf(c[v] + fc1b[row]);
      }
    }
  }
  __syncthreads();

  // Phase 2: gx_b = Y @ W_ih^T  (M=64, N=1536, K=256).
  // Wave w owns ids w*48..w*48+47 -> mt = w/2 constant per wave.
  {
    __bf16* gxb = gx + (size_t)b * NWS * G3H;
    int mt = wave >> 1;
    bf16x16 afr[8];                      // hoisted A fragments from LDS (K=256)
#pragma unroll
    for (int k = 0; k < 8; ++k)
      afr[k] = load_a_frag(&Y[mt * 16][k * 32], II, lane);  // ds_load
    for (int i = 0; i < 48; ++i) {
      int nt = (wave * 48 + i) % 96;
      f32x8 c = {};
#pragma unroll
      for (int k = 0; k < 8; ++k)
        c = wmma_bf16(afr[k], load_b_frag(Wih + nt * 16 * II + k * 32, II, lane), c);
      int col = nt * 16 + n;
      float bias = bih[col];
#pragma unroll
      for (int v = 0; v < 8; ++v) {
        int row = mt * 16 + half * 8 + v;
        gxb[(size_t)row * G3H + col] = f2bf(c[v] + bias);
      }
    }
  }
}

// ---------------------------------------------------------------------------
// Kernel: GRU scan. 32 WGs x 16 batch rows; h lives in LDS across 64 steps.
//   gh[16,1536] = h_bf16 @ W_hh^T + b_hh   (WMMA, A-frags hoisted from LDS)
//   gates in f32 (paired/vectorized), last 24 steps stream h to global (bf16).
// ---------------------------------------------------------------------------
__global__ __launch_bounds__(256) void gru_scan(
    const __bf16* __restrict__ gx,    // [B][64][1536]
    const __bf16* __restrict__ Whh,   // [1536][512] (N-major B)
    const float*  __restrict__ bhh,   // [1536]
    __bf16* __restrict__ hseq)        // [B][24][512]
{
  __shared__ __align__(32) float  hF[16][HH];    // 32 KB  (f32 master copy)
  __shared__ __align__(32) __bf16 hB[16][HH];    // 16 KB  (WMMA A source)
  __shared__ __align__(32) float  gh[16][G3H];   // 96 KB
  int blk = blockIdx.x;
  int tid = threadIdx.x, wave = tid >> 5, lane = tid & 31;
  int n = lane & 15, half = lane >> 4;

  for (int i = tid; i < 16 * HH; i += 256) {
    ((float*)hF)[i]  = 0.0f;
    ((__bf16*)hB)[i] = f2bf(0.0f);
  }
  __syncthreads();

  for (int t = 0; t < NWS; ++t) {
    // Hoist the 16 A-fragments of h (M=16, K=512) once per step.
    bf16x16 afr[16];
#pragma unroll
    for (int k = 0; k < 16; ++k)
      afr[k] = load_a_frag(&hB[0][k * 32], HH, lane);       // ds_load

    // gh = h @ W_hh^T + b_hh : N=1536 -> 96 tiles, 12 per wave
    for (int i = 0; i < 12; ++i) {
      int nt = wave + i * 8;            // 0..95
      if (i + 1 < 12) {                 // prefetch next streamed B tile (L2-hot)
        __builtin_prefetch(Whh + (size_t)(nt + 8) * 16 * HH + (size_t)n * HH +
                               half * 256, 0, 1);
      }
      f32x8 c = {};
#pragma unroll
      for (int k = 0; k < 16; ++k)
        c = wmma_bf16(afr[k],
                      load_b_frag(Whh + (size_t)nt * 16 * HH + k * 32, HH, lane), c);
      int col = nt * 16 + n;
      float bias = bhh[col];
#pragma unroll
      for (int v = 0; v < 8; ++v) gh[half * 8 + v][col] = c[v] + bias;
    }
    __syncthreads();

    // Gate update: 16 rows x 512 h-elems, 2 consecutive elems per thread.
#pragma unroll 2
    for (int j = 0; j < 16; ++j) {
      int idx = (tid + j * 256) * 2;          // 0..8190, even
      int row = idx >> 9, hh = idx & (HH - 1);
      const __bf16* gp = gx + (((size_t)(blk * 16 + row) * NWS + t) * G3H);
      bf16x2 vr = *(const bf16x2*)(gp + hh);
      bf16x2 vz = *(const bf16x2*)(gp + HH + hh);
      bf16x2 vn = *(const bf16x2*)(gp + 2 * HH + hh);
      f32x2 ghr = *(const f32x2*)&gh[row][hh];
      f32x2 ghz = *(const f32x2*)&gh[row][HH + hh];
      f32x2 ghn = *(const f32x2*)&gh[row][2 * HH + hh];
      f32x2 hp  = *(const f32x2*)&hF[row][hh];
      f32x2 hnew;
      bf16x2 hbn;
#pragma unroll
      for (int e = 0; e < 2; ++e) {
        float r  = sigmoid_f(bf2f(vr[e]) + ghr[e]);
        float z  = sigmoid_f(bf2f(vz[e]) + ghz[e]);
        float nn = tanh_f(bf2f(vn[e]) + r * ghn[e]);
        hnew[e]  = (1.0f - z) * nn + z * hp[e];
        hbn[e]   = f2bf(hnew[e]);
      }
      *(f32x2*)&hF[row][hh]  = hnew;
      *(bf16x2*)&hB[row][hh] = hbn;
      if (t >= NWS - PRE) {
        *(bf16x2*)&hseq[(((size_t)(blk * 16 + row)) * PRE + (t - (NWS - PRE))) * HH +
                        hh] = hbn;
      }
    }
    __syncthreads();
  }
}

// ---------------------------------------------------------------------------
// Kernel: fc2 epilogue. Rows = (b,t') flattened: 512*24 = 12288.
//   out[12288,256] = hseq[12288,512] @ fc2_w^T + fc2_b   (f32 output)
// ---------------------------------------------------------------------------
__global__ __launch_bounds__(256) void fc2_kernel(
    const __bf16* __restrict__ hseq,  // [12288][512]
    const __bf16* __restrict__ W2,    // [256][512] (N-major B)
    const float*  __restrict__ b2,    // [256]
    float* __restrict__ out)          // [12288][256]
{
  int blk = blockIdx.x;               // 192 blocks x 64 rows
  int tid = threadIdx.x, wave = tid >> 5, lane = tid & 31;
  int n = lane & 15, half = lane >> 4;
  int row0 = blk * 64;
  int mt = wave >> 1;                 // constant per wave (ids w*8..w*8+7)
  bf16x16 afr[16];                    // hoisted A fragments (K=512)
#pragma unroll
  for (int k = 0; k < 16; ++k)
    afr[k] = load_a_frag(hseq + ((size_t)(row0 + mt * 16)) * HH + k * 32, HH, lane);
  for (int i = 0; i < 8; ++i) {
    int nt = (wave * 8 + i) & 15;
    f32x8 c = {};
#pragma unroll
    for (int k = 0; k < 16; ++k)
      c = wmma_bf16(afr[k], load_b_frag(W2 + nt * 16 * HH + k * 32, HH, lane), c);
    int col = nt * 16 + n;
    float bias = b2[col];
#pragma unroll
    for (int v = 0; v < 8; ++v)
      out[((size_t)(row0 + mt * 16 + half * 8 + v)) * II + col] = c[v] + bias;
  }
}

// ---------------------------------------------------------------------------
// Host entry
// ---------------------------------------------------------------------------
extern "C" void kernel_launch(void* const* d_in, const int* in_sizes, int n_in,
                              void* d_out, int out_size, void* d_ws, size_t ws_size,
                              hipStream_t stream) {
  const float* x    = (const float*)d_in[0];
  const float* fc1w = (const float*)d_in[1];
  const float* fc1b = (const float*)d_in[2];
  const float* Wih  = (const float*)d_in[3];
  const float* Whh  = (const float*)d_in[4];
  const float* bih  = (const float*)d_in[5];
  const float* bhh  = (const float*)d_in[6];
  const float* fc2w = (const float*)d_in[7];
  const float* fc2b = (const float*)d_in[8];
  float* out = (float*)d_out;

  char* ws = (char*)d_ws;
  size_t off = 0;
  auto alloc = [&](size_t bytes) -> char* {
    char* p = ws + off;
    off += (bytes + 255) & ~(size_t)255;
    return p;
  };
  __bf16* xT_bf   = (__bf16*)alloc((size_t)BB * II * WS * 2);     //  33.6 MB
  __bf16* fc1_bf  = (__bf16*)alloc((size_t)NWS * WS * 2);         //  16 KB
  __bf16* Wih_bf  = (__bf16*)alloc((size_t)G3H * II * 2);         //  768 KB
  __bf16* Whh_bf  = (__bf16*)alloc((size_t)G3H * HH * 2);         //  1.5 MB
  __bf16* fc2_bf  = (__bf16*)alloc((size_t)II * HH * 2);          //  256 KB
  __bf16* gx_bf   = (__bf16*)alloc((size_t)BB * NWS * G3H * 2);   // 100.7 MB
  __bf16* hseq_bf = (__bf16*)alloc((size_t)BB * PRE * HH * 2);    //  12.6 MB

  // Weight converts (tiny; element counts are all even)
  cvt_f32_bf16<<<(NWS * WS / 2 + 255) / 256, 256, 0, stream>>>(fc1w, fc1_bf,
                                                               NWS * WS);
  cvt_f32_bf16<<<(G3H * II / 2 + 255) / 256, 256, 0, stream>>>(Wih, Wih_bf,
                                                               G3H * II);
  cvt_f32_bf16<<<(G3H * HH / 2 + 255) / 256, 256, 0, stream>>>(Whh, Whh_bf,
                                                               G3H * HH);
  cvt_f32_bf16<<<(II * HH / 2 + 255) / 256, 256, 0, stream>>>(fc2w, fc2_bf,
                                                              II * HH);

  // x -> xT (bf16, N-major for the fc1 GEMM)
  transpose_x<<<dim3(BB, WS / 32, II / 32), 256, 0, stream>>>(x, xT_bf);

  // Fused fc1 + gate-input projection (one WG per batch)
  fused_fc1_gx<<<BB, 256, 0, stream>>>(xT_bf, fc1_bf, fc1b, Wih_bf, bih, gx_bf);

  // GRU recurrence: 32 WGs x 16 batch rows, h resident in LDS
  gru_scan<<<BB / 16, 256, 0, stream>>>(gx_bf, Whh_bf, bhh, hseq_bf);

  // fc2 epilogue over the last 24 timesteps
  fc2_kernel<<<(BB * PRE) / 64, 256, 0, stream>>>(hseq_bf, fc2_bf, fc2b, out);
}
